// GELU128_39857296507236
// MI455X (gfx1250) — matrix operand order repaired and checked by reading, not compiled, not run on hardware
//
#include <hip/hip_runtime.h>

typedef __attribute__((ext_vector_type(16))) __bf16 v16bf;
typedef __attribute__((ext_vector_type(8)))  __bf16 v8bf;
typedef __attribute__((ext_vector_type(8)))  float  v8f;

#define B_  8
#define T_  2048
#define D_  1024
#define SCALE_ 0.03125f   // 1/sqrt(1024)

union AFrag { v16bf v; v8bf h[2]; };

// ---------------------------------------------------------------------------
// Phase 1: S = causal_mask(x x^T * scale) per 16-row Q tile, softmax in LDS,
// write P (bf16, zero-padded to 64-key chunks) to workspace.
// S row-block (16 x T fp32 = 128KB) lives in dynamic LDS.
// Two accumulators break the WMMA RAW chain on C.
// ---------------------------------------------------------------------------
__global__ __launch_bounds__(256) void attn_softmax_kernel(
    const float* __restrict__ x, __bf16* __restrict__ P)
{
  __shared__ __align__(16) __bf16 Qs[16 * D_];   // 32 KB bf16 Q tile
  extern __shared__ float Srow[];                // 16 * T_ fp32 (dynamic, 128 KB)

  const int tid = threadIdx.x;
  const int b   = blockIdx.x / (T_ / 16);
  const int qt  = blockIdx.x % (T_ / 16);
  const int q0  = qt * 16;

  // stage Q tile fp32 -> bf16 (coalesced)
  const float* xq = x + ((size_t)(b * T_ + q0)) * D_;
  for (int i = tid; i < 16 * D_; i += 256) Qs[i] = (__bf16)xq[i];
  __syncthreads();

  const int lane = tid & 31;
  const int wv   = tid >> 5;     // 8 waves
  const int hi   = lane >> 4;    // 0/1 half of wave
  const int lo16 = lane & 15;
  const int ncol = q0 + 16;      // causal extent (tile-aligned)
  const int nct  = ncol >> 4;    // 16-wide key tiles to compute

  for (int ct = wv; ct < nct; ct += 8) {
    const int s0 = ct * 16;
    v8f acc0 = {}, acc1 = {};
    const float* xk = x + ((size_t)(b * T_ + s0 + lo16)) * D_;  // B operand key row
    const __bf16* qrow = Qs + lo16 * D_;
    for (int dc = 0; dc < D_; dc += 64) {
      // A = Q 16x32 bf16 fragments: K = e + hi*16
      AFrag a0, a1;
      a0.h[0] = *(const v8bf*)(qrow + dc + hi * 16);
      a0.h[1] = *(const v8bf*)(qrow + dc + 8 + hi * 16);
      a1.h[0] = *(const v8bf*)(qrow + dc + 32 + hi * 16);
      a1.h[1] = *(const v8bf*)(qrow + dc + 40 + hi * 16);
      // B = K^T 32x16 bf16: lane col = key s0+lo16, K rows d = dc(+32) + hi*16 + e
      const float4* k0 = (const float4*)(xk + dc + hi * 16);
      const float4* k1 = (const float4*)(xk + dc + 32 + hi * 16);
      float4 f0 = k0[0], f1 = k0[1], f2 = k1[0], f3 = k1[1];
      v16bf b0, b1;
      b0[0]  = (__bf16)f0.x; b0[1]  = (__bf16)f0.y; b0[2]  = (__bf16)f0.z; b0[3]  = (__bf16)f0.w;
      b0[4]  = (__bf16)f1.x; b0[5]  = (__bf16)f1.y; b0[6]  = (__bf16)f1.z; b0[7]  = (__bf16)f1.w;
      b1[0]  = (__bf16)f2.x; b1[1]  = (__bf16)f2.y; b1[2]  = (__bf16)f2.z; b1[3]  = (__bf16)f2.w;
      b1[4]  = (__bf16)f3.x; b1[5]  = (__bf16)f3.y; b1[6]  = (__bf16)f3.z; b1[7]  = (__bf16)f3.w;
      float4 f4 = k0[2], f5 = k0[3], f6 = k1[2], f7 = k1[3];
      b0[8]  = (__bf16)f4.x; b0[9]  = (__bf16)f4.y; b0[10] = (__bf16)f4.z; b0[11] = (__bf16)f4.w;
      b0[12] = (__bf16)f5.x; b0[13] = (__bf16)f5.y; b0[14] = (__bf16)f5.z; b0[15] = (__bf16)f5.w;
      b1[8]  = (__bf16)f6.x; b1[9]  = (__bf16)f6.y; b1[10] = (__bf16)f6.z; b1[11] = (__bf16)f6.w;
      b1[12] = (__bf16)f7.x; b1[13] = (__bf16)f7.y; b1[14] = (__bf16)f7.z; b1[15] = (__bf16)f7.w;
      acc0 = __builtin_amdgcn_wmma_f32_16x16x32_bf16(false, a0.v, false, b0,
                                                     (short)0, acc0, false, false);
      acc1 = __builtin_amdgcn_wmma_f32_16x16x32_bf16(false, a1.v, false, b1,
                                                     (short)0, acc1, false, false);
    }
    const v8f acc = acc0 + acc1;
    // C layout: element r -> row M = r + 8*hi, col N = lo16. Scale + strict mask.
#pragma unroll
    for (int r = 0; r < 8; ++r) {
      const int m = r + hi * 8;
      const int s = s0 + lo16;
      Srow[m * T_ + s] = (s >= q0 + m) ? -INFINITY : acc[r] * SCALE_;
    }
  }
  __syncthreads();

  // NaN-safe softmax: 16 threads per row (shfl width 16 stays in-wave halves)
  const int m = tid >> 4;
  const int j = tid & 15;
  const float* Sm = Srow + m * T_;
  float mx = -INFINITY;
  for (int s = j; s < ncol; s += 16) mx = fmaxf(mx, Sm[s]);
  for (int k = 1; k < 16; k <<= 1) mx = fmaxf(mx, __shfl_xor(mx, k, 16));
  const float msafe = (mx == -INFINITY) ? 0.0f : mx;
  float sum = 0.0f;
  for (int s = j; s < ncol; s += 16) sum += __expf(Sm[s] - msafe);
  for (int k = 1; k < 16; k <<= 1) sum += __shfl_xor(sum, k, 16);
  const float inv = (sum > 0.0f) ? 1.0f / sum : 0.0f;
  const int ncolp = (ncol + 63) & ~63;    // zero-pad to K=64 chunk for phase 2
  __bf16* Pr = P + ((size_t)(b * T_ + q0 + m)) * T_;
  for (int s = j; s < ncolp; s += 16) {
    float e = (s < ncol) ? __expf(Sm[s] - msafe) * inv : 0.0f;
    Pr[s] = (__bf16)e;
  }
}

// ---------------------------------------------------------------------------
// Phase 2: context = P @ x. Per round: stage P(16x64) + x chunk transposed
// (Xbt[d][k], padded rows -> 16B-aligned b128 DS loads, ~2-way conflicts).
// 8 waves per block each own a 16-wide D tile; two WMMAs (two accumulators)
// per barrier round. fp32 context written into d_out.
// ---------------------------------------------------------------------------
#define XROW_ 72   // 64 keys + 8 pad elements (16B) per d-row

__global__ __launch_bounds__(256) void attn_pv_kernel(
    const float* __restrict__ x, const __bf16* __restrict__ P,
    float* __restrict__ ctx)
{
  __shared__ __align__(16) __bf16 Pa[16 * 64];       // 2 KB
  __shared__ __align__(16) __bf16 Xbt[128 * XROW_];  // 18 KB, [d][k] transposed

  const int tid = threadIdx.x;
  const int b   = blockIdx.x / (T_ / 16);
  const int qt  = blockIdx.x % (T_ / 16);
  const int q0  = qt * 16;
  const int d0b = blockIdx.y * 128;
  const int lane = tid & 31, wv = tid >> 5;
  const int hi = lane >> 4, lo16 = lane & 15;
  const int ncolp = ((q0 + 16) + 63) & ~63;

  v8f acc0 = {}, acc1 = {};
  const int dloc = wv * 16 + lo16;
  for (int kc = 0; kc < ncolp; kc += 64) {
    for (int i = tid; i < 16 * 64; i += 256) {      // P rows, coalesced
      const int r = i >> 6, c = i & 63;
      Pa[i] = P[((size_t)(b * T_ + q0 + r)) * T_ + kc + c];
    }
    for (int i = tid; i < 64 * 128; i += 256) {     // x chunk, transpose into LDS
      const int k = i >> 7, c = i & 127;            // global read coalesced over c
      Xbt[c * XROW_ + k] = (__bf16)x[((size_t)(b * T_ + kc + k)) * D_ + d0b + c];
    }
    __syncthreads();
    // A = P 16x64 (two 16x32 fragments), B = x 64x16 (two 32x16 fragments)
    AFrag a0, a1, b0, b1;
    const __bf16* prow = Pa + lo16 * 64;
    a0.h[0] = *(const v8bf*)(prow + hi * 16);
    a0.h[1] = *(const v8bf*)(prow + 8 + hi * 16);
    a1.h[0] = *(const v8bf*)(prow + 32 + hi * 16);
    a1.h[1] = *(const v8bf*)(prow + 40 + hi * 16);
    const __bf16* xcol = Xbt + dloc * XROW_ + hi * 16;   // K rows hi*16+e
    b0.h[0] = *(const v8bf*)(xcol);
    b0.h[1] = *(const v8bf*)(xcol + 8);
    b1.h[0] = *(const v8bf*)(xcol + 32);
    b1.h[1] = *(const v8bf*)(xcol + 40);
    acc0 = __builtin_amdgcn_wmma_f32_16x16x32_bf16(false, a0.v, false, b0.v,
                                                   (short)0, acc0, false, false);
    acc1 = __builtin_amdgcn_wmma_f32_16x16x32_bf16(false, a1.v, false, b1.v,
                                                   (short)0, acc1, false, false);
    __syncthreads();
  }
  const v8f acc = acc0 + acc1;
#pragma unroll
  for (int r = 0; r < 8; ++r) {
    const int mm = r + hi * 8;
    ctx[((size_t)(b * T_ + q0 + mm)) * D_ + d0b + dloc] = acc[r];
  }
}

// ---------------------------------------------------------------------------
// Phase 3: per-row cosine similarity -> novelty gate -> tanh-GELU, in place
// over d_out (context). Reads of context complete before writes (same block).
// ---------------------------------------------------------------------------
__global__ __launch_bounds__(256) void gate_gelu_kernel(
    const float* __restrict__ x, const float* __restrict__ la_p,
    const float* __restrict__ ls_p, float* __restrict__ out)
{
  __shared__ float red[3][8];
  const int row = blockIdx.x;       // b*T + t
  const int tid = threadIdx.x;
  const float* xr = x + (size_t)row * D_;
  const float* cr = out + (size_t)row * D_;
  float sxx = 0.f, scc = 0.f, sxc = 0.f;
  for (int d = tid; d < D_; d += 256) {
    const float xv = xr[d], cv = cr[d];
    sxx += xv * xv; scc += cv * cv; sxc += xv * cv;
  }
  for (int k = 1; k < 32; k <<= 1) {
    sxx += __shfl_xor(sxx, k, 32);
    scc += __shfl_xor(scc, k, 32);
    sxc += __shfl_xor(sxc, k, 32);
  }
  const int lane = tid & 31, wv = tid >> 5;
  if (lane == 0) { red[0][wv] = sxx; red[1][wv] = scc; red[2][wv] = sxc; }
  __syncthreads();
  if (tid == 0) {
    float a = 0.f, b2 = 0.f, c = 0.f;
    for (int i = 0; i < 8; ++i) { a += red[0][i]; b2 += red[1][i]; c += red[2][i]; }
    red[0][0] = a; red[1][0] = b2; red[2][0] = c;
  }
  __syncthreads();
  sxx = red[0][0]; scc = red[1][0]; sxc = red[2][0];
  const float nx = fmaxf(sqrtf(sxx), 1e-12f);
  const float nc = fmaxf(sqrtf(scc), 1e-12f);
  const float cosv = sxc / (nx * nc);
  float nov = fminf(fmaxf(1.0f - cosv, 0.0f), 2.0f) * 0.5f;
  if (!(nov == nov)) nov = 1.0f;     // nan_to_num(nan=1.0)
  const float alpha = log1pf(__expf(la_p[0]));
  const float sigma = log1pf(__expf(ls_p[0]));
  const float gate  = 1.0f + alpha * tanhf(sigma * nov);
  for (int d = tid; d < D_; d += 256) {
    const float g = xr[d] * gate;
    const float u = 0.7978845608028654f * (g + 0.044715f * g * g * g);
    out[(size_t)row * D_ + d] = 0.5f * g * (1.0f + tanhf(u));
  }
}

// ---------------------------------------------------------------------------
extern "C" void kernel_launch(void* const* d_in, const int* in_sizes, int n_in,
                              void* d_out, int out_size, void* d_ws, size_t ws_size,
                              hipStream_t stream)
{
  const float* x  = (const float*)d_in[0];
  const float* la = (const float*)d_in[1];
  const float* ls = (const float*)d_in[2];
  float* out = (float*)d_out;
  __bf16* P  = (__bf16*)d_ws;    // B*T*T bf16 = 64 MB attention probabilities

  // 128 KB dynamic LDS for the fp32 score row-block (CDNA5 WGP: up to 320 KB)
  (void)hipFuncSetAttribute((const void*)attn_softmax_kernel,
                            hipFuncAttributeMaxDynamicSharedMemorySize,
                            16 * T_ * (int)sizeof(float));

  dim3 g1(B_ * (T_ / 16));
  attn_softmax_kernel<<<g1, 256, 16 * T_ * sizeof(float), stream>>>(x, P);

  dim3 g2(B_ * (T_ / 16), D_ / 128);
  attn_pv_kernel<<<g2, 256, 0, stream>>>(x, P, out);

  dim3 g3(B_ * T_);
  gate_gelu_kernel<<<g3, 256, 0, stream>>>(x, la, ls, out);
}
